// RoPEALiBiMultiheadAttention_39453569581362
// MI455X (gfx1250) — matrix operand order repaired and checked by reading, not compiled, not run on hardware
//
#include <hip/hip_runtime.h>
#include <hip/hip_bf16.h>

// ---------------------------------------------------------------------------
// Types
// ---------------------------------------------------------------------------
typedef __bf16 bf16;
typedef __attribute__((ext_vector_type(4)))  float  v4f;
typedef __attribute__((ext_vector_type(8)))  float  v8f;
typedef __attribute__((ext_vector_type(8)))  __bf16 v8bf;
typedef __attribute__((ext_vector_type(16))) __bf16 v16bf;

#define D_MODEL   1024
#define NUM_HEADS 16
#define HEAD_DIM  64
#define BATCH     2
#define SEQ       2048

static __device__ __forceinline__ v8f wmma_bf16(v16bf a, v16bf b, v8f c) {
    // D = A(16x32 bf16) x B(32x16 bf16) + C(16x16 f32)
    return __builtin_amdgcn_wmma_f32_16x16x32_bf16(
        false, a, false, b, (short)0, c, false, false);
}

// A-matrix (16x32, 16-bit) per-lane K packing (ISA 7.12.2):
// lanes 0-15: i=0..7 -> K=i,      i=8..15 -> K=16+(i-8)
// lanes16-31: i=0..7 -> K=8+i,    i=8..15 -> K=24+(i-8)
static __device__ __forceinline__ v16bf load_a_frag(const bf16* rowp, int half, int koff) {
    v8bf lo = *(const v8bf*)(rowp + koff + 8 * half);
    v8bf hi = *(const v8bf*)(rowp + koff + 16 + 8 * half);
    v16bf r;
#pragma unroll
    for (int i = 0; i < 8; ++i) { r[i] = lo[i]; r[8 + i] = hi[i]; }
    return r;
}

// B-matrix (32x16, 16-bit): same K packing as A (lanes 0-15: K=0..7,16..23; lanes 16-31: K=8..15,24..31).
static __device__ __forceinline__ v16bf load_b_frag(const bf16* rowp, int half, int koff) {
    v8bf lo = *(const v8bf*)(rowp + koff + 8 * half);
    v8bf hi = *(const v8bf*)(rowp + koff + 16 + 8 * half);
    v16bf r;
#pragma unroll
    for (int i = 0; i < 8; ++i) { r[i] = lo[i]; r[8 + i] = hi[i]; }
    return r;
}

// 16 contiguous elements -> bf16, written as two 16B LDS stores.
static __device__ __forceinline__ void load_conv16(const float* __restrict__ src,
                                                   bf16* dst) {
    const v4f* s4 = (const v4f*)src;
    v4f f0 = s4[0], f1 = s4[1], f2 = s4[2], f3 = s4[3];
    v8bf p0, p1;
#pragma unroll
    for (int i = 0; i < 4; ++i) {
        p0[i]     = (bf16)f0[i];
        p0[4 + i] = (bf16)f1[i];
        p1[i]     = (bf16)f2[i];
        p1[4 + i] = (bf16)f3[i];
    }
    *(v8bf*)dst       = p0;
    *(v8bf*)(dst + 8) = p1;
}
static __device__ __forceinline__ void load_conv16(const bf16* __restrict__ src,
                                                   bf16* dst) {
    const v8bf* s8 = (const v8bf*)src;
    *(v8bf*)dst       = s8[0];
    *(v8bf*)(dst + 8) = s8[1];
}

// ---------------------------------------------------------------------------
// GEMM: Y[M,N] = A[M,K] * W[N,K]^T + bias[N]   (Y fp32, W fp32, A fp32|bf16)
// Block: 256 thr (8 waves, 4x2). Tile: BM=128, BN=128, BK=32.
// Wave (wm,wn) -> rows 32*wm..+32, cols 64*wn..+64  => 8 WMMA / K-step.
// ---------------------------------------------------------------------------
template <typename TA>
__global__ __launch_bounds__(256) void gemm_bias_kernel(
    const TA* __restrict__ A, const float* __restrict__ W,
    const float* __restrict__ bias, float* __restrict__ Y,
    int M, int N, int K)
{
    __shared__ alignas(16) bf16 a_s[128][40];  // 80B rows (16B multiple)
    __shared__ alignas(16) bf16 w_s[128][40];

    const int tid  = threadIdx.x;
    const int wv   = tid >> 5;
    const int lane = tid & 31;
    const int half = lane >> 4;
    const int lr   = lane & 15;
    const int wm   = wv >> 1;      // 0..3
    const int wn   = wv & 1;       // 0..1

    const int nb = blockIdx.x * 128;
    const int mb = blockIdx.y * 128;

    const int lrow = tid >> 1;            // 0..127
    const int lc0  = (tid & 1) * 16;      // 0 or 16

    v8f zero = {};
    v8f acc[2][4];
#pragma unroll
    for (int am = 0; am < 2; ++am)
#pragma unroll
        for (int nt = 0; nt < 4; ++nt) acc[am][nt] = zero;

    for (int k0 = 0; k0 < K; k0 += 32) {
        load_conv16(A + (size_t)(mb + lrow) * K + k0 + lc0, &a_s[lrow][lc0]);
        load_conv16(W + (size_t)(nb + lrow) * K + k0 + lc0, &w_s[lrow][lc0]);
        if (k0 + 32 < K) {
            __builtin_prefetch((const void*)(A + (size_t)(mb + lrow) * K + k0 + 32 + lc0), 0, 3);
            __builtin_prefetch((const void*)(W + (size_t)(nb + lrow) * K + k0 + 32 + lc0), 0, 3);
        }
        __syncthreads();

        v16bf av0 = load_a_frag(&a_s[32 * wm + lr][0], half, 0);
        v16bf av1 = load_a_frag(&a_s[32 * wm + 16 + lr][0], half, 0);
#pragma unroll
        for (int nt = 0; nt < 4; ++nt) {
            v16bf bv = load_b_frag(&w_s[64 * wn + 16 * nt + lr][0], half, 0);
            acc[0][nt] = wmma_bf16(av0, bv, acc[0][nt]);
            acc[1][nt] = wmma_bf16(av1, bv, acc[1][nt]);
        }
        __syncthreads();
    }

    // C/D layout: col = lr, row = r + 8*half
#pragma unroll
    for (int nt = 0; nt < 4; ++nt) {
        const int ng = nb + 64 * wn + 16 * nt + lr;
        const float bvv = bias[ng];
#pragma unroll
        for (int am = 0; am < 2; ++am)
#pragma unroll
            for (int r = 0; r < 8; ++r) {
                const int mg = mb + 32 * wm + 16 * am + r + 8 * half;
                Y[(size_t)mg * N + ng] = acc[am][nt][r] + bvv;
            }
    }
}

// ---------------------------------------------------------------------------
// RoPE + reorder: fp32 [B,L,H*64] -> bf16 [B,H,L,64], optional RoPE.
// ---------------------------------------------------------------------------
__global__ __launch_bounds__(256) void rope_reorder_kernel(
    const float* __restrict__ x, const float* __restrict__ pe,
    bf16* __restrict__ out, int L, int H, int do_rope)
{
    const int bl = blockIdx.x;
    const int b = bl / L, l = bl % L;
    const float* xr  = x + (size_t)bl * D_MODEL;
    const float* per = pe + (size_t)l * HEAD_DIM;

    for (int idx = threadIdx.x; idx < D_MODEL; idx += 256) {
        const int h = idx >> 6, d = idx & 63;
        float val;
        if (do_rope) {
            if (d < 32) {
                val = xr[h * 64 + 2 * d] * per[2 * d] - xr[h * 64 + 2 * d + 1] * per[2 * d + 1];
            } else {
                const int d2 = d - 32;
                val = xr[h * 64 + 2 * d2] * per[2 * d2 + 1] + xr[h * 64 + 2 * d2 + 1] * per[2 * d2];
            }
        } else {
            val = xr[idx];
        }
        out[(((size_t)b * H + h) * L + l) * HEAD_DIM + d] = (bf16)val;
    }
}

// ---------------------------------------------------------------------------
// Flash attention: per block (64-query tile, one (b,h)); 4 waves, 128 threads.
// ---------------------------------------------------------------------------
__global__ __launch_bounds__(128) void flash_attn_kernel(
    const bf16* __restrict__ Q, const bf16* __restrict__ Kt,
    const bf16* __restrict__ V, const float* __restrict__ alibi,
    bf16* __restrict__ Out, int H, int L, float scaling)
{
    __shared__ alignas(16) bf16 q_s[64][72];      // [q row][d]
    __shared__ alignas(16) bf16 k_s[64][72];      // [key][d]
    __shared__ alignas(16) bf16 vt_s[64][72];     // transposed: [d][key]
    __shared__ alignas(16) bf16 p_s[4][16][72];   // per-wave P: [q row][key]

    const int bh = blockIdx.y;
    const int b = bh / H, h = bh % H;
    const int q0 = blockIdx.x * 64;

    const int tid  = threadIdx.x;
    const int wv   = tid >> 5;
    const int lane = tid & 31;
    const int half = lane >> 4;
    const int lr   = lane & 15;

    const int lrow = tid >> 1;            // 0..63
    const int lc0  = (tid & 1) * 32;      // 0 or 32

    // Load Q tile (64x64): 32 contiguous bf16/thread, vectorized
    {
        const bf16* qg = Q + ((size_t)bh * L + q0) * HEAD_DIM + lrow * HEAD_DIM + lc0;
        const v8bf* s = (const v8bf*)qg;
#pragma unroll
        for (int i = 0; i < 4; ++i) *(v8bf*)&q_s[lrow][lc0 + 8 * i] = s[i];
    }
    __syncthreads();

    const v16bf aq0 = load_a_frag(&q_s[16 * wv + lr][0], half, 0);
    const v16bf aq1 = load_a_frag(&q_s[16 * wv + lr][0], half, 32);

    float m_st[8], l_st[8];
#pragma unroll
    for (int r = 0; r < 8; ++r) { m_st[r] = -3.0e38f; l_st[r] = 0.0f; }
    v8f zero = {};
    v8f oacc[4];
#pragma unroll
    for (int dt = 0; dt < 4; ++dt) oacc[dt] = zero;

    const bf16* kgb = Kt + (size_t)bh * L * HEAD_DIM;
    const bf16* vgb = V  + (size_t)bh * L * HEAD_DIM;

    for (int kt = 0; kt < L / 64; ++kt) {
        __syncthreads();  // previous tile fully consumed
        {
            const v8bf* ks = (const v8bf*)(kgb + (kt * 64 + lrow) * HEAD_DIM + lc0);
            const v8bf* vs = (const v8bf*)(vgb + (kt * 64 + lrow) * HEAD_DIM + lc0);
            v8bf kv0 = ks[0], kv1 = ks[1], kv2 = ks[2], kv3 = ks[3];
            v8bf vv0 = vs[0], vv1 = vs[1], vv2 = vs[2], vv3 = vs[3];
            *(v8bf*)&k_s[lrow][lc0]      = kv0;
            *(v8bf*)&k_s[lrow][lc0 + 8]  = kv1;
            *(v8bf*)&k_s[lrow][lc0 + 16] = kv2;
            *(v8bf*)&k_s[lrow][lc0 + 24] = kv3;
#pragma unroll
            for (int i = 0; i < 8; ++i) {
                vt_s[lc0 + i][lrow]      = vv0[i];
                vt_s[lc0 + 8 + i][lrow]  = vv1[i];
                vt_s[lc0 + 16 + i][lrow] = vv2[i];
                vt_s[lc0 + 24 + i][lrow] = vv3[i];
            }
        }
        if (kt + 1 < L / 64) {  // hide next tile's HBM->L2 latency under 16 WMMAs
            __builtin_prefetch((const void*)(kgb + ((kt + 1) * 64 + lrow) * HEAD_DIM + lc0), 0, 3);
            __builtin_prefetch((const void*)(vgb + ((kt + 1) * 64 + lrow) * HEAD_DIM + lc0), 0, 3);
        }
        __syncthreads();

        // ---- S = scale * Q K^T + alibi ----
        float sv[4][8];
#pragma unroll
        for (int nt = 0; nt < 4; ++nt) {
            v8f s = zero;
            v16bf bk0 = load_b_frag(&k_s[16 * nt + lr][0], half, 0);
            v16bf bk1 = load_b_frag(&k_s[16 * nt + lr][0], half, 32);
            s = wmma_bf16(aq0, bk0, s);
            s = wmma_bf16(aq1, bk1, s);
            const int key_g = kt * 64 + 16 * nt + lr;
#pragma unroll
            for (int r = 0; r < 8; ++r) {
                const int row_g = q0 + 16 * wv + r + 8 * half;
                sv[nt][r] = s[r] * scaling +
                            alibi[((size_t)h * L + row_g) * L + key_g];
            }
        }

        // ---- online softmax: row = (r, half); row's 16 cols live in one 16-lane half ----
        float mnew[8], alpha[8];
#pragma unroll
        for (int r = 0; r < 8; ++r) {
            float mx = fmaxf(fmaxf(sv[0][r], sv[1][r]), fmaxf(sv[2][r], sv[3][r]));
            mx = fmaxf(mx, __shfl_xor(mx, 1, 32));
            mx = fmaxf(mx, __shfl_xor(mx, 2, 32));
            mx = fmaxf(mx, __shfl_xor(mx, 4, 32));
            mx = fmaxf(mx, __shfl_xor(mx, 8, 32));
            mnew[r]  = fmaxf(m_st[r], mx);
            alpha[r] = __expf(m_st[r] - mnew[r]);
        }
#pragma unroll
        for (int nt = 0; nt < 4; ++nt)
#pragma unroll
            for (int r = 0; r < 8; ++r)
                sv[nt][r] = __expf(sv[nt][r] - mnew[r]);
#pragma unroll
        for (int r = 0; r < 8; ++r) {
            float s4 = sv[0][r] + sv[1][r] + sv[2][r] + sv[3][r];
            s4 += __shfl_xor(s4, 1, 32);
            s4 += __shfl_xor(s4, 2, 32);
            s4 += __shfl_xor(s4, 4, 32);
            s4 += __shfl_xor(s4, 8, 32);
            l_st[r] = l_st[r] * alpha[r] + s4;
            m_st[r] = mnew[r];
        }
#pragma unroll
        for (int dt = 0; dt < 4; ++dt)
#pragma unroll
            for (int r = 0; r < 8; ++r)
                oacc[dt][r] *= alpha[r];

        // ---- P: C-layout -> per-wave LDS -> A-layout (wave-ordered, no barrier) ----
#pragma unroll
        for (int nt = 0; nt < 4; ++nt)
#pragma unroll
            for (int r = 0; r < 8; ++r)
                p_s[wv][r + 8 * half][16 * nt + lr] = (bf16)sv[nt][r];

        v16bf ap0 = load_a_frag(&p_s[wv][lr][0], half, 0);
        v16bf ap1 = load_a_frag(&p_s[wv][lr][0], half, 32);

        // ---- O += P V ----
#pragma unroll
        for (int dt = 0; dt < 4; ++dt) {
            v16bf bv0 = load_b_frag(&vt_s[16 * dt + lr][0], half, 0);
            v16bf bv1 = load_b_frag(&vt_s[16 * dt + lr][0], half, 32);
            oacc[dt] = wmma_bf16(ap0, bv0, oacc[dt]);
            oacc[dt] = wmma_bf16(ap1, bv1, oacc[dt]);
        }
    }

    // ---- normalize + write bf16 [B, L, H*64] ----
    float inv[8];
#pragma unroll
    for (int r = 0; r < 8; ++r) inv[r] = 1.0f / l_st[r];
#pragma unroll
    for (int dt = 0; dt < 4; ++dt)
#pragma unroll
        for (int r = 0; r < 8; ++r) {
            const int row_g = q0 + 16 * wv + r + 8 * half;
            Out[((size_t)b * L + row_g) * D_MODEL + h * HEAD_DIM + 16 * dt + lr] =
                (bf16)(oacc[dt][r] * inv[r]);
        }
}

// ---------------------------------------------------------------------------
// Launch
// ---------------------------------------------------------------------------
extern "C" void kernel_launch(void* const* d_in, const int* in_sizes, int n_in,
                              void* d_out, int out_size, void* d_ws, size_t ws_size,
                              hipStream_t stream)
{
    (void)in_sizes; (void)n_in; (void)out_size; (void)ws_size;

    const float* query  = (const float*)d_in[0];
    const float* keyx   = (const float*)d_in[1];
    const float* value  = (const float*)d_in[2];
    const float* alibi  = (const float*)d_in[3];
    const float* pos    = (const float*)d_in[4];
    const float* Wq = (const float*)d_in[5];  const float* bq = (const float*)d_in[6];
    const float* Wk = (const float*)d_in[7];  const float* bk = (const float*)d_in[8];
    const float* Wv = (const float*)d_in[9];  const float* bv = (const float*)d_in[10];
    const float* Wo = (const float*)d_in[11]; const float* bo = (const float*)d_in[12];

    const int M = BATCH * SEQ;           // 4096
    const int N = D_MODEL, K = D_MODEL;  // 1024

    // Workspace layout (72 MB):
    float* qf = (float*)d_ws;                 // 16 MB fp32 [B,L,D]
    float* kf = qf + (size_t)M * D_MODEL;     // 16 MB
    float* vf = kf + (size_t)M * D_MODEL;     // 16 MB
    bf16*  qh = (bf16*)(vf + (size_t)M * D_MODEL);  // 8 MB bf16 [B,H,L,64]
    bf16*  kh = qh + (size_t)M * D_MODEL;           // 8 MB
    bf16*  vh = kh + (size_t)M * D_MODEL;           // 8 MB
    bf16*  attn = (bf16*)qf;  // alias: q-proj fp32 dead after its RoPE pass

    dim3 gBlk(256), gGrid(N / 128, M / 128);

    gemm_bias_kernel<float><<<gGrid, gBlk, 0, stream>>>(query, Wq, bq, qf, M, N, K);
    gemm_bias_kernel<float><<<gGrid, gBlk, 0, stream>>>(keyx,  Wk, bk, kf, M, N, K);
    gemm_bias_kernel<float><<<gGrid, gBlk, 0, stream>>>(value, Wv, bv, vf, M, N, K);

    rope_reorder_kernel<<<dim3(M), dim3(256), 0, stream>>>(qf, pos, qh, SEQ, NUM_HEADS, 1);
    rope_reorder_kernel<<<dim3(M), dim3(256), 0, stream>>>(kf, pos, kh, SEQ, NUM_HEADS, 1);
    rope_reorder_kernel<<<dim3(M), dim3(256), 0, stream>>>(vf, pos, vh, SEQ, NUM_HEADS, 0);

    flash_attn_kernel<<<dim3(SEQ / 64, BATCH * NUM_HEADS), dim3(128), 0, stream>>>(
        qh, kh, vh, alibi, attn, NUM_HEADS, SEQ, 0.125f);

    gemm_bias_kernel<bf16><<<gGrid, gBlk, 0, stream>>>(attn, Wo, bo, (float*)d_out, M, N, K);
}
